// SingleHead_Self_Attention_34832184771186
// MI455X (gfx1250) — matrix operand order repaired and checked by reading, not compiled, not run on hardware
//
#include <hip/hip_runtime.h>
#include <hip/hip_bf16.h>

#define N_TOK 8192
#define DIM   1024

typedef __attribute__((ext_vector_type(16))) __bf16 v16bf;
typedef __attribute__((ext_vector_type(8)))  __bf16 v8bf;
typedef __attribute__((ext_vector_type(4)))  __bf16 v4bf;
typedef __attribute__((ext_vector_type(4)))  float  v4f;
typedef __attribute__((ext_vector_type(8)))  float  v8f;

// ---------------- WMMA fragment loaders (wave32, 16x16x32 bf16) ----------------
// A-matrix 16x32 (MxK): lanes 0-15 -> M=lane, K = k0+0..15 (elems 0..15);
// lanes 16-31 -> M=lane-16, K = k0+16..31.
static __device__ __forceinline__ v16bf load_a_lds(const __bf16* As, int ldk, int kk, int lane) {
    const int m    = lane & 15;
    const int koff = (lane >> 4) << 4;              // 0 or 16
    const __bf16* p = As + m * ldk + kk + koff;
    v8bf lo = *(const v8bf*)p;                      // ds_load_b128
    v8bf hi = *(const v8bf*)(p + 8);
    v16bf a;
#pragma unroll
    for (int i = 0; i < 8; ++i) { a[i] = lo[i]; a[i + 8] = hi[i]; }
    return a;
}

// B-matrix 32x16 (KxN): lanes 0-15 -> N=lane, K=k0..k0+15; lanes 16-31 -> K=k0+16..31.
// Bt is the K-major source (Bt[n][k] == B[k][n]) so each lane reads 16 contiguous bf16.
static __device__ __forceinline__ v16bf load_b_bf16(const __bf16* __restrict__ Bt,
                                                    size_t ldb, int col0, int k0, int lane) {
    const int n    = col0 + (lane & 15);
    const int koff = (lane >> 4) << 4;              // 0 or 16
    return *(const v16bf*)(Bt + (size_t)n * ldb + (size_t)(k0 + koff));
}

// Cooperative copy of a rows x kwidth bf16 strip into LDS (8 halves per chunk).
static __device__ __forceinline__ void fill_strip_bf16(__bf16* dst, const __bf16* __restrict__ src,
                                                       size_t lda, int row0, int kwidth,
                                                       int tid, int nthreads) {
    const int chunks = 16 * kwidth / 8;
    const int ck     = kwidth / 8;
    for (int c = tid; c < chunks; c += nthreads) {
        const int r  = c / ck;
        const int kk = (c % ck) * 8;
        *(v8bf*)(dst + r * kwidth + kk) =
            *(const v8bf*)(src + (size_t)(row0 + r) * lda + kk);
    }
}

// C/D 16x16 f32: VGPR r -> row = r (+8 for lanes 16-31), col = lane%16.

// ---------------- fp32 -> bf16 conversion (grid-stride) ----------------
__global__ void cvt_f32_bf16(const float* __restrict__ src, __bf16* __restrict__ dst, size_t n) {
    size_t i = (size_t)blockIdx.x * blockDim.x + threadIdx.x;
    const size_t stride = (size_t)gridDim.x * blockDim.x;
    for (; i < n; i += stride) dst[i] = (__bf16)src[i];
}

// ---------------- QKV projections: Y = X @ W^T + b ----------------
// Block: 8 waves x 64 cols = 512 cols; grid (N/16, D/512, 3).
// A strip (16 x 1024 of Xbf, 32 KB) staged in LDS, shared by all 8 waves.
__global__ void qkv_gemm(const __bf16* __restrict__ Xbf, const __bf16* __restrict__ Wbf,
                         const float* __restrict__ bq, const float* __restrict__ bk,
                         const float* __restrict__ bv,
                         __bf16* __restrict__ Qbf, __bf16* __restrict__ Kbf,
                         __bf16* __restrict__ Vt) {
    __shared__ __bf16 As[16 * DIM];                 // 32 KB of 320 KB/WGP
    const int lane = threadIdx.x & 31;
    const int wave = threadIdx.x >> 5;
    const int row0 = blockIdx.x * 16;
    const int colw = blockIdx.y * 512 + wave * 64;
    const int mat  = blockIdx.z;
    const __bf16* W    = Wbf + (size_t)mat * DIM * DIM;   // row n of W == column n of B
    const float*  bias = (mat == 0) ? bq : (mat == 1) ? bk : bv;

    fill_strip_bf16(As, Xbf, DIM, row0, DIM, threadIdx.x, blockDim.x);

    v8f c[4];
#pragma unroll
    for (int j = 0; j < 4; ++j) {
        const float bval = bias[colw + j * 16 + (lane & 15)];
#pragma unroll
        for (int r = 0; r < 8; ++r) c[j][r] = bval;
    }
    __syncthreads();

    for (int k0 = 0; k0 < DIM; k0 += 32) {
        v16bf a = load_a_lds(As, DIM, k0, lane);
        // prefetch next k-slab of the B stream (global_prefetch_b8)
        if (k0 + 64 < DIM)
            __builtin_prefetch(W + (size_t)(colw + (lane & 15)) * DIM + k0 + 64, 0, 3);
#pragma unroll
        for (int j = 0; j < 4; ++j) {
            v16bf b = load_b_bf16(W, DIM, colw + j * 16, k0, lane);
            c[j] = __builtin_amdgcn_wmma_f32_16x16x32_bf16(false, a, false, b,
                                                           (short)0, c[j], false, false);
        }
    }

    const int rbase = row0 + ((lane >> 4) << 3);
#pragma unroll
    for (int j = 0; j < 4; ++j) {
        const int col = colw + j * 16 + (lane & 15);
        if (mat == 2) {                       // V: store transposed -> 8 contiguous bf16
            v8bf vv;
#pragma unroll
            for (int r = 0; r < 8; ++r) vv[r] = (__bf16)c[j][r];
            *(v8bf*)(Vt + (size_t)col * N_TOK + rbase) = vv;
        } else {
            __bf16* Dst = (mat == 0) ? Qbf : Kbf;
#pragma unroll
            for (int r = 0; r < 8; ++r)
                Dst[(size_t)(rbase + r) * DIM + col] = (__bf16)c[j][r];
        }
    }
}

// ---------------- sim = (1 - Q K^T) / sqrt(D) (fp32 out) ----------------
// Block: 8 waves x 64 cols = 512 cols; grid (N/16, N/512).
// A strip (16 rows of Q, 32 KB bf16) staged in LDS once.
__global__ void sim_gemm(const __bf16* __restrict__ Qbf, const __bf16* __restrict__ Kbf,
                         const int* __restrict__ indicator, float* __restrict__ S) {
    __shared__ __bf16 As[16 * DIM];
    const int lane = threadIdx.x & 31;
    const int wave = threadIdx.x >> 5;
    const int row0 = blockIdx.x * 16;
    const int colw = blockIdx.y * 512 + wave * 64;

    fill_strip_bf16(As, Qbf, DIM, row0, DIM, threadIdx.x, blockDim.x);

    v8f c[4];
#pragma unroll
    for (int j = 0; j < 4; ++j)
#pragma unroll
        for (int r = 0; r < 8; ++r) c[j][r] = 0.0f;
    __syncthreads();

    for (int k0 = 0; k0 < DIM; k0 += 32) {
        v16bf a = load_a_lds(As, DIM, k0, lane);
        if (k0 + 64 < DIM)
            __builtin_prefetch(Kbf + (size_t)(colw + (lane & 15)) * DIM + k0 + 64, 0, 3);
#pragma unroll
        for (int j = 0; j < 4; ++j) {
            v16bf b = load_b_bf16(Kbf, DIM, colw + j * 16, k0, lane);  // K row n == col n of K^T
            c[j] = __builtin_amdgcn_wmma_f32_16x16x32_bf16(false, a, false, b,
                                                           (short)0, c[j], false, false);
        }
    }

    const float scale = 0.03125f;                 // 1/sqrt(1024)
    const bool  inv   = (*indicator == 0);
    const int   rbase = row0 + ((lane >> 4) << 3);
#pragma unroll
    for (int j = 0; j < 4; ++j) {
        const int col = colw + j * 16 + (lane & 15);
#pragma unroll
        for (int r = 0; r < 8; ++r) {
            const float s = inv ? (1.0f - c[j][r]) * scale : c[j][r] * scale;
            S[(size_t)(rbase + r) * N_TOK + col] = s;
        }
    }
}

// ---------------- softmax over axis=0: per-column online max + reciprocal-sum --------
__global__ void col_softmax_stats(const float* __restrict__ S,
                                  float* __restrict__ cmax, float* __restrict__ crsum) {
    const int j = blockIdx.x * blockDim.x + threadIdx.x;   // one column per thread
    float m = -3.402823466e38f, sum = 0.0f;
    for (int i = 0; i < N_TOK; ++i) {
        const float s  = S[(size_t)i * N_TOK + j];
        const float nm = fmaxf(m, s);
        sum = sum * __expf(m - nm) + __expf(s - nm);
        m = nm;
    }
    cmax[j]  = m;
    crsum[j] = 1.0f / sum;                                 // reciprocal -> multiply later
}

__global__ void softmax_norm(float* __restrict__ S, const float* __restrict__ cmax,
                             const float* __restrict__ crsum) {
    const size_t total  = (size_t)N_TOK * N_TOK;
    const size_t stride = (size_t)gridDim.x * blockDim.x;
    for (size_t i = (size_t)blockIdx.x * blockDim.x + threadIdx.x; i < total; i += stride) {
        const int j = (int)(i & (N_TOK - 1));              // column index
        S[i] = __expf(S[i] - cmax[j]) * crsum[j];
    }
}

// ---------------- attn_out = attn_w @ V ----------------
// Block: 16 waves x 64 cols = all 1024 cols -> attn_w row-strip read once per block.
// attn_w (fp32, HBM-resident) is staged + converted to bf16 in LDS per 256-wide k-slab,
// so the conversion happens once per block instead of once per wave.
#define AV_SLAB 256
__global__ void av_gemm(const float* __restrict__ Wmat, const __bf16* __restrict__ Vt,
                        float* __restrict__ Out) {
    __shared__ __bf16 Aw[16 * AV_SLAB];           // 8 KB
    const int lane = threadIdx.x & 31;
    const int wave = threadIdx.x >> 5;            // 0..15
    const int row0 = blockIdx.x * 16;
    const int colw = wave * 64;

    v8f c[4];
#pragma unroll
    for (int j = 0; j < 4; ++j)
#pragma unroll
        for (int r = 0; r < 8; ++r) c[j][r] = 0.0f;

    for (int k0 = 0; k0 < N_TOK; k0 += AV_SLAB) {
        // Cooperative fill: 16 x 256 fp32 -> bf16 in LDS (4 floats per chunk).
        {
            const int chunks = 16 * AV_SLAB / 4;  // 1024
            for (int cn = threadIdx.x; cn < chunks; cn += blockDim.x) {
                const int r  = cn >> 6;           // / (AV_SLAB/4)
                const int kk = (cn & 63) << 2;
                v4f f = *(const v4f*)(Wmat + (size_t)(row0 + r) * N_TOK + k0 + kk);
                v4bf h;
#pragma unroll
                for (int t = 0; t < 4; ++t) h[t] = (__bf16)f[t];
                *(v4bf*)(Aw + r * AV_SLAB + kk) = h;
            }
        }
        // Prefetch next attn_w slab while this one is consumed.
        if (k0 + AV_SLAB < N_TOK)
            __builtin_prefetch(Wmat + (size_t)(row0 + (threadIdx.x & 15)) * N_TOK +
                               k0 + AV_SLAB + ((threadIdx.x >> 4) << 3), 0, 3);
        __syncthreads();

        for (int kk = 0; kk < AV_SLAB; kk += 32) {
            v16bf a = load_a_lds(Aw, AV_SLAB, kk, lane);
#pragma unroll
            for (int j = 0; j < 4; ++j) {
                v16bf b = load_b_bf16(Vt, N_TOK, colw + j * 16, k0 + kk, lane);
                c[j] = __builtin_amdgcn_wmma_f32_16x16x32_bf16(false, a, false, b,
                                                               (short)0, c[j], false, false);
            }
        }
        __syncthreads();
    }

    const int rbase = row0 + ((lane >> 4) << 3);
#pragma unroll
    for (int j = 0; j < 4; ++j) {
        const int col = colw + j * 16 + (lane & 15);
#pragma unroll
        for (int r = 0; r < 8; ++r)
            Out[(size_t)(rbase + r) * DIM + col] = c[j][r];
    }
}

// ---------------- host-side launcher ----------------
extern "C" void kernel_launch(void* const* d_in, const int* in_sizes, int n_in,
                              void* d_out, int out_size, void* d_ws, size_t ws_size,
                              hipStream_t stream) {
    (void)in_sizes; (void)n_in; (void)out_size; (void)ws_size;

    const float* X   = (const float*)d_in[0];
    const int*   ind = (const int*)d_in[1];
    const float* Wq  = (const float*)d_in[2];
    const float* bq  = (const float*)d_in[3];
    const float* Wk  = (const float*)d_in[4];
    const float* bk  = (const float*)d_in[5];
    const float* Wv  = (const float*)d_in[6];
    const float* bv  = (const float*)d_in[7];

    float* attn_out = (float*)d_out;                         // [N, D]
    float* S        = attn_out + (size_t)N_TOK * DIM;        // [N, N] scores -> attn_w (in place)

    // Workspace carve-up (256B aligned): ~70 MB total.
    char*  ws  = (char*)d_ws;
    size_t off = 0;
    auto carve = [&](size_t bytes) -> void* {
        void* p = ws + off;
        off += (bytes + 255) & ~(size_t)255;
        return p;
    };
    __bf16* Xbf  = (__bf16*)carve((size_t)N_TOK * DIM * 2);
    __bf16* Wbf  = (__bf16*)carve((size_t)3 * DIM * DIM * 2);
    __bf16* Qbf  = (__bf16*)carve((size_t)N_TOK * DIM * 2);
    __bf16* Kbf  = (__bf16*)carve((size_t)N_TOK * DIM * 2);
    __bf16* Vt   = (__bf16*)carve((size_t)DIM * N_TOK * 2);
    float*  cmax = (float*)carve((size_t)N_TOK * 4);
    float*  crs  = (float*)carve((size_t)N_TOK * 4);

    // 1) bf16 conversions of X and the three weight matrices.
    cvt_f32_bf16<<<1024, 256, 0, stream>>>(X,  Xbf, (size_t)N_TOK * DIM);
    cvt_f32_bf16<<<256, 256, 0, stream>>>(Wq, Wbf,                         (size_t)DIM * DIM);
    cvt_f32_bf16<<<256, 256, 0, stream>>>(Wk, Wbf + (size_t)DIM * DIM,     (size_t)DIM * DIM);
    cvt_f32_bf16<<<256, 256, 0, stream>>>(Wv, Wbf + (size_t)2 * DIM * DIM, (size_t)DIM * DIM);

    // 2) Q/K/V projections (V stored transposed as Vt).
    qkv_gemm<<<dim3(N_TOK / 16, DIM / 512, 3), 256, 0, stream>>>(
        Xbf, Wbf, bq, bk, bv, Qbf, Kbf, Vt);

    // 3) Scores S = (1 - Q K^T) / sqrt(D)  (indicator read on device).
    sim_gemm<<<dim3(N_TOK / 16, N_TOK / 512), 256, 0, stream>>>(Qbf, Kbf, ind, S);

    // 4) Column-wise softmax (axis=0): stats pass, then in-place normalize.
    col_softmax_stats<<<N_TOK / 256, 256, 0, stream>>>(S, cmax, crs);
    softmax_norm<<<8192, 256, 0, stream>>>(S, cmax, crs);

    // 5) attn_out = attn_w @ V.
    av_gemm<<<N_TOK / 16, 512, 0, stream>>>(S, Vt, attn_out);
}
